// Attention_49778670961018
// MI455X (gfx1250) — matrix operand order repaired and checked by reading, not compiled, not run on hardware
//
#include <hip/hip_runtime.h>

typedef __bf16 bf16;
typedef bf16  v16bf __attribute__((ext_vector_type(16)));
typedef bf16  v8bf  __attribute__((ext_vector_type(8)));
typedef float v8f   __attribute__((ext_vector_type(8)));

static constexpr int H_  = 1024;   // hidden
static constexpr int A_  = 1024;   // attention out
static constexpr int NS_ = 8192;   // attendee stmts
static constexpr int NE_ = 4096;   // attendee eres
static constexpr int M_  = 8192;   // attenders

// ---------------- WMMA fragment loaders (ISA 7.12.2 bf16 layouts) ----------------

// A-matrix 16x32 bf16: lane m (0..15) holds K {0..7,16..23}; lane m+16 holds K {8..15,24..31}
__device__ __forceinline__ v16bf load_frag_A(const bf16* __restrict__ base, int ld,
                                             int m0, int k0, int lane) {
  const int r  = lane & 15;
  const int kh = (lane >> 4) << 3;                 // 0 or 8
  const bf16* p = base + (size_t)(m0 + r) * ld + k0 + kh;
  v8bf lo = *(const v8bf*)(p);                     // K = kh .. kh+7
  v8bf hi = *(const v8bf*)(p + 16);                // K = kh+16 .. kh+23
  return __builtin_shufflevector(lo, hi, 0,1,2,3,4,5,6,7,8,9,10,11,12,13,14,15);
}

// B-matrix 32x16 bf16 from Bt (row-major [N,K]): lane n holds col n, K {0..7,16..23}; lane n+16: K {8..15,24..31}
__device__ __forceinline__ v16bf load_frag_Bt(const bf16* __restrict__ base, int ld,
                                              int n0, int k0, int lane) {
  const int n  = lane & 15;
  const int kh = (lane >> 4) << 3;                 // 0 or 8
  const bf16* p = base + (size_t)(n0 + n) * ld + k0 + kh;
  v8bf lo = *(const v8bf*)(p);                     // K = kh .. kh+7
  v8bf hi = *(const v8bf*)(p + 16);                // K = kh+16 .. kh+23
  return __builtin_shufflevector(lo, hi, 0,1,2,3,4,5,6,7,8,9,10,11,12,13,14,15);
}

// ---------------- GEMM: C[M,N] = A[M,K] x Bt[N,K]^T (+bias)(+tanh) ----------------
// Block: 256 threads = 8 waves arranged 4(M) x 2(N); wave tile 32x64 (2x4 WMMA frags).
// Block tile 128x128. Per K-step: 6 fragment loads feed 8 back-to-back WMMAs.
// EPI: 0 = f32 out; 1 = bf16 out + bias; 2 = bf16 out; 3 = f32 out + bias + tanh
template<int EPI>
__global__ __launch_bounds__(256)
void gemm_wmma_bf16(const bf16* __restrict__ Am, int lda,
                    const bf16* __restrict__ Bm, int ldb,
                    const float* __restrict__ bias,
                    void* __restrict__ Cm, int ldc, int K) {
  const int lane = threadIdx.x & 31;
  const int wave = threadIdx.x >> 5;
  const int m0 = blockIdx.y * 128 + (wave >> 1) * 32;   // 4 waves down M
  const int n0 = blockIdx.x * 128 + (wave & 1) * 64;    // 2 waves across N

  v8f acc[2][4];
  #pragma unroll
  for (int i = 0; i < 2; ++i)
    #pragma unroll
    for (int j = 0; j < 4; ++j)
      acc[i][j] = (v8f){0.f, 0.f, 0.f, 0.f, 0.f, 0.f, 0.f, 0.f};

  for (int k0 = 0; k0 < K; k0 += 32) {
    v16bf a0 = load_frag_A(Am, lda, m0,      k0, lane);
    v16bf a1 = load_frag_A(Am, lda, m0 + 16, k0, lane);
    v16bf b0 = load_frag_Bt(Bm, ldb, n0,      k0, lane);
    v16bf b1 = load_frag_Bt(Bm, ldb, n0 + 16, k0, lane);
    v16bf b2 = load_frag_Bt(Bm, ldb, n0 + 32, k0, lane);
    v16bf b3 = load_frag_Bt(Bm, ldb, n0 + 48, k0, lane);
    // pull the next K-slab toward the WGP (gfx1250 global_prefetch_b8)
    __builtin_prefetch(Am + (size_t)(m0 + (lane & 15)) * lda + k0 + 32, 0, 1);
    __builtin_prefetch(Bm + (size_t)(n0 + (lane & 15)) * ldb + k0 + 32, 0, 1);

    acc[0][0] = __builtin_amdgcn_wmma_f32_16x16x32_bf16(false, a0, false, b0, (short)0, acc[0][0], false, false);
    acc[0][1] = __builtin_amdgcn_wmma_f32_16x16x32_bf16(false, a0, false, b1, (short)0, acc[0][1], false, false);
    acc[0][2] = __builtin_amdgcn_wmma_f32_16x16x32_bf16(false, a0, false, b2, (short)0, acc[0][2], false, false);
    acc[0][3] = __builtin_amdgcn_wmma_f32_16x16x32_bf16(false, a0, false, b3, (short)0, acc[0][3], false, false);
    acc[1][0] = __builtin_amdgcn_wmma_f32_16x16x32_bf16(false, a1, false, b0, (short)0, acc[1][0], false, false);
    acc[1][1] = __builtin_amdgcn_wmma_f32_16x16x32_bf16(false, a1, false, b1, (short)0, acc[1][1], false, false);
    acc[1][2] = __builtin_amdgcn_wmma_f32_16x16x32_bf16(false, a1, false, b2, (short)0, acc[1][2], false, false);
    acc[1][3] = __builtin_amdgcn_wmma_f32_16x16x32_bf16(false, a1, false, b3, (short)0, acc[1][3], false, false);
  }

  // C/D layout: lane<16 -> N=lane, M=r; lane>=16 -> N=lane-16, M=r+8
  const int n  = lane & 15;
  const int mo = (lane >> 4) << 3;
  #pragma unroll
  for (int i = 0; i < 2; ++i) {
    #pragma unroll
    for (int j = 0; j < 4; ++j) {
      const int nn = n0 + j * 16 + n;
      const float bv = (EPI == 1 || EPI == 3) ? bias[nn] : 0.f;
      #pragma unroll
      for (int r = 0; r < 8; ++r) {
        float v = acc[i][j][r] + bv;
        if (EPI == 3) v = tanhf(v);
        const size_t idx = (size_t)(m0 + i * 16 + mo + r) * ldc + nn;
        if (EPI == 1 || EPI == 2) ((bf16*)Cm)[idx]  = (bf16)v;
        else                      ((float*)Cm)[idx] = v;
      }
    }
  }
}

// ---------------- Row softmax, f32 in-place -> bf16 (halves pass-2 bytes) ----------------
template<int L>
__global__ __launch_bounds__(256)
void softmax_row_inplace_bf16(float* __restrict__ S) {
  constexpr int T  = 256;
  constexpr int PT = L / T;
  float* row  = S + (size_t)blockIdx.x * L;
  bf16*  orow = (bf16*)row;                        // bf16 row aliases first half of f32 row
  const int t = threadIdx.x;

  float v[PT];
  float mx = -__builtin_inff();
  #pragma unroll
  for (int i = 0; i < PT; ++i) { v[i] = row[t + i * T]; mx = fmaxf(mx, v[i]); }

  __shared__ float red[T];
  red[t] = mx; __syncthreads();
  for (int s = T / 2; s > 0; s >>= 1) { if (t < s) red[t] = fmaxf(red[t], red[t + s]); __syncthreads(); }
  mx = red[0]; __syncthreads();

  float sum = 0.f;
  #pragma unroll
  for (int i = 0; i < PT; ++i) { v[i] = expf(v[i] - mx); sum += v[i]; }
  red[t] = sum; __syncthreads();
  for (int s = T / 2; s > 0; s >>= 1) { if (t < s) red[t] += red[t + s]; __syncthreads(); }
  const float inv = 1.f / red[0];
  __syncthreads();                                  // all f32 reads retired before aliased writes
  #pragma unroll
  for (int i = 0; i < PT; ++i) orow[t + i * T] = (bf16)(v[i] * inv);
}

// ---------------- f32 -> bf16 converters ----------------
__global__ void cvt_f32_bf16(const float* __restrict__ s, bf16* __restrict__ d, size_t n) {
  size_t i = (size_t)blockIdx.x * blockDim.x + threadIdx.x;
  if (i < n) d[i] = (bf16)s[i];
}
__global__ void cvt_rows_f32_bf16(const float* __restrict__ s, int sld,
                                  bf16* __restrict__ d, int dld, int cols) {
  int c = blockIdx.x * blockDim.x + threadIdx.x;
  int r = blockIdx.y;
  if (c < cols) d[(size_t)r * dld + c] = (bf16)s[(size_t)r * sld + c];
}

// ---------------- f32 [R,C] -> bf16 transposed [C,R] (LDS 32x32 tile) ----------------
__global__ __launch_bounds__(256)
void transpose_f32_bf16(const float* __restrict__ src, int R, int C,
                        bf16* __restrict__ dst) {
  __shared__ float tile[32][33];
  const int bx = blockIdx.x * 32;                  // column base (C)
  const int by = blockIdx.y * 32;                  // row base (R)
  const int tx = threadIdx.x & 31;
  const int ty = threadIdx.x >> 5;                 // 0..7
  #pragma unroll
  for (int i = 0; i < 32; i += 8)
    tile[ty + i][tx] = src[(size_t)(by + ty + i) * C + bx + tx];
  __syncthreads();
  #pragma unroll
  for (int i = 0; i < 32; i += 8)
    dst[(size_t)(bx + ty + i) * R + by + tx] = (bf16)tile[tx][ty + i];
}

// ---------------- driver ----------------
extern "C" void kernel_launch(void* const* d_in, const int* in_sizes, int n_in,
                              void* d_out, int out_size, void* d_ws, size_t ws_size,
                              hipStream_t stream) {
  const float* f_As   = (const float*)d_in[0];   // [NS,H]
  const float* f_Ae   = (const float*)d_in[1];   // [NE,H]
  const float* f_att  = (const float*)d_in[2];   // [M,H]
  const float* f_Wss  = (const float*)d_in[3];   // [H,H]
  const float* b_ss   = (const float*)d_in[4];
  const float* f_Wes  = (const float*)d_in[5];   // [H,H]
  const float* b_es   = (const float*)d_in[6];
  const float* f_Wlin = (const float*)d_in[7];   // [A,3H]
  const float* b_lin  = (const float*)d_in[8];
  float* out = (float*)d_out;                    // [M,A] f32

  char* ws = (char*)d_ws;
  size_t off = 0;
  auto alloc = [&](size_t bytes) -> void* {
    void* p = ws + off;
    off = (off + bytes + 255) & ~(size_t)255;
    return p;
  };
  bf16* As_bf   = (bf16*)alloc((size_t)NS_ * H_ * 2);       // attendee stmts bf16 [NS,H]
  bf16* Ae_bf   = (bf16*)alloc((size_t)NE_ * H_ * 2);       // attendee eres bf16 [NE,H]
  bf16* AsT_bf  = (bf16*)alloc((size_t)H_ * NS_ * 2);       // transposed stmts  [H,NS]
  bf16* AeT_bf  = (bf16*)alloc((size_t)H_ * NE_ * 2);       // transposed eres   [H,NE]
  bf16* Wss_bf  = (bf16*)alloc((size_t)H_ * H_ * 2);
  bf16* Wes_bf  = (bf16*)alloc((size_t)H_ * H_ * 2);
  bf16* Wlin_bf = (bf16*)alloc((size_t)A_ * 3 * H_ * 2);
  bf16* Pss_bf  = (bf16*)alloc((size_t)NS_ * H_ * 2);       // projected stmts
  bf16* Pes_bf  = (bf16*)alloc((size_t)NE_ * H_ * 2);       // projected eres
  bf16* cat_bf  = (bf16*)alloc((size_t)M_ * 3 * H_ * 2);    // [att | ctx_ss | ctx_es]
  float* Sss    = (float*)alloc((size_t)M_ * NS_ * 4);      // scores^T ss (f32 -> bf16 in place)
  float* Ses    = (float*)alloc((size_t)M_ * NE_ * 4);      // scores^T es

  const int TB = 256;
  auto cgrid = [&](size_t n) { return dim3((unsigned)((n + TB - 1) / TB)); };

  // 1) downconvert operands to bf16 (attender straight into cat[:, 0:H]) + transposes
  cvt_f32_bf16<<<cgrid((size_t)NS_ * H_), TB, 0, stream>>>(f_As,  As_bf,  (size_t)NS_ * H_);
  cvt_f32_bf16<<<cgrid((size_t)NE_ * H_), TB, 0, stream>>>(f_Ae,  Ae_bf,  (size_t)NE_ * H_);
  cvt_f32_bf16<<<cgrid((size_t)H_ * H_),  TB, 0, stream>>>(f_Wss, Wss_bf, (size_t)H_ * H_);
  cvt_f32_bf16<<<cgrid((size_t)H_ * H_),  TB, 0, stream>>>(f_Wes, Wes_bf, (size_t)H_ * H_);
  cvt_f32_bf16<<<cgrid((size_t)A_ * 3 * H_), TB, 0, stream>>>(f_Wlin, Wlin_bf, (size_t)A_ * 3 * H_);
  cvt_rows_f32_bf16<<<dim3(H_ / TB, M_), TB, 0, stream>>>(f_att, H_, cat_bf, 3 * H_, H_);
  transpose_f32_bf16<<<dim3(H_ / 32, NS_ / 32), TB, 0, stream>>>(f_As, NS_, H_, AsT_bf);
  transpose_f32_bf16<<<dim3(H_ / 32, NE_ / 32), TB, 0, stream>>>(f_Ae, NE_, H_, AeT_bf);

  // 2) projections: P = X @ W^T + b  (bf16 out)
  gemm_wmma_bf16<1><<<dim3(H_ / 128, NS_ / 128), TB, 0, stream>>>(
      As_bf, H_, Wss_bf, H_, b_ss, Pss_bf, H_, H_);
  gemm_wmma_bf16<1><<<dim3(H_ / 128, NE_ / 128), TB, 0, stream>>>(
      Ae_bf, H_, Wes_bf, H_, b_es, Pes_bf, H_, H_);

  // 3) transposed scores: St[m,n] = att[m,:] . P[n,:]  (f32 out)
  gemm_wmma_bf16<0><<<dim3(NS_ / 128, M_ / 128), TB, 0, stream>>>(
      cat_bf, 3 * H_, Pss_bf, H_, nullptr, Sss, NS_, H_);
  gemm_wmma_bf16<0><<<dim3(NE_ / 128, M_ / 128), TB, 0, stream>>>(
      cat_bf, 3 * H_, Pes_bf, H_, nullptr, Ses, NE_, H_);

  // 4) softmax over attendee dim = row softmax of St; bf16 weights written in place
  softmax_row_inplace_bf16<NS_><<<dim3(M_), TB, 0, stream>>>(Sss);
  softmax_row_inplace_bf16<NE_><<<dim3(M_), TB, 0, stream>>>(Ses);

  // 5) contexts: ctx[m,h] = sum_n w[m,n] * attendee[n,h] = W @ (att^T)^T
  //    -> cat[:, H:2H] / cat[:, 2H:3H]; Bt = pre-transposed attendees (fast contiguous path)
  gemm_wmma_bf16<2><<<dim3(H_ / 128, M_ / 128), TB, 0, stream>>>(
      (const bf16*)Sss, 2 * NS_, AsT_bf, NS_, nullptr, cat_bf + H_,     3 * H_, NS_);
  gemm_wmma_bf16<2><<<dim3(H_ / 128, M_ / 128), TB, 0, stream>>>(
      (const bf16*)Ses, 2 * NE_, AeT_bf, NE_, nullptr, cat_bf + 2 * H_, 3 * H_, NE_);

  // 6) out = tanh(cat @ W_lin^T + b_lin)  (f32 out)
  gemm_wmma_bf16<3><<<dim3(A_ / 128, M_ / 128), TB, 0, stream>>>(
      cat_bf, 3 * H_, Wlin_bf, 3 * H_, b_lin, out, A_, 3 * H_);
}
